// TransformerConvBlock_80942953660875
// MI455X (gfx1250) — compile-verified, matrix operand order checked by reading, not run on hardware
//
#include <hip/hip_runtime.h>

// ---------------------------------------------------------------------------
// MI455X (gfx1250): GEMMs via v_wmma_f32_16x16x32_f16, LDS double-buffered
// with GLOBAL_LOAD_ASYNC_TO_LDS_B128 (ASYNCcnt) when available.
// f32 softmax / LayerNorm / residual. wave32 everywhere.
// ---------------------------------------------------------------------------

typedef _Float16 v8h  __attribute__((ext_vector_type(8)));
typedef _Float16 v16h __attribute__((ext_vector_type(16)));
typedef float    v8f  __attribute__((ext_vector_type(8)));
typedef int      vi4  __attribute__((__vector_size__(4 * sizeof(int))));

#define BT    8
#define CH    288
#define HH    48
#define WW    48
#define NHEAD 9
#define HD    32
#define CFF   2592
#define NT    (BT*HH*WW)   // 18432 tokens

// ---- async-to-LDS availability (device pass only) ----
#if defined(__HIP_DEVICE_COMPILE__) && __has_builtin(__builtin_amdgcn_global_load_async_to_lds_b128)
#define ASYNC_OK 1
#else
#define ASYNC_OK 0
#endif

#if ASYNC_OK
// builtin signature (probe-verified): arg0 is AS(1) v4i*, arg1 AS(3) v4i*
#define AS1_V4I(p) ((__attribute__((address_space(1))) vi4*)(p))
#define AS3_V4I(p) ((__attribute__((address_space(3))) vi4*)(p))
#if __has_builtin(__builtin_amdgcn_s_wait_asynccnt)
#define WAIT_ASYNC(n) __builtin_amdgcn_s_wait_asynccnt(n)
#else
#define WAIT_ASYNC(n) asm volatile("s_wait_asynccnt %0" :: "i"(n) : "memory")
#endif
#endif

// ---- workspace layout (bytes) ----
#define O_XLH   ((size_t)0)                       // f16 [NT][288]
#define O_WQKV  ((size_t)10616832)                // f16 [864][288]
#define O_WO    ((size_t)11114496)                // f16 [288][288]
#define O_W1    ((size_t)11280384)                // f16 [2592][288]
#define O_W2    ((size_t)12773376)                // f16 [288][2592]
#define O_QKV   ((size_t)14266368)                // f16 [NT][864]   (reused: h_in f16 [NT][288])
#define O_ATTN  ((size_t)46116864)                // f16 [NT][288]
#define O_Y     ((size_t)56733696)                // f32 [NT][288]   (reused: y2)
#define O_H1    ((size_t)77967360)                // f16 [NT][2592]
#define O_STAT  ((size_t)173518848)               // f32 [NT][2]

// ---------------------------------------------------------------------------
// NCHW f32 -> token-major f16 (LDS tile transpose, coalesced both ways)
// ---------------------------------------------------------------------------
__global__ __launch_bounds__(256) void k_xl(const float* __restrict__ x,
                                            _Float16* __restrict__ xlh) {
  __shared__ _Float16 xt[WW][CH + 8];
  int blk = blockIdx.x;          // over B*H
  int b = blk / HH, i = blk % HH;
  const float* xs = x + (size_t)b * CH * (HH*WW) + (size_t)i * WW;
  for (int idx = threadIdx.x; idx < CH*WW; idx += 256) {
    int c = idx / WW, j = idx % WW;
    xt[j][c] = (_Float16)xs[(size_t)c * (HH*WW) + j];
  }
  __syncthreads();
  _Float16* od = xlh + (size_t)(b*HH + i) * WW * CH;
  for (int idx = threadIdx.x; idx < WW*CH; idx += 256) {
    int j = idx / CH, c = idx % CH;
    od[(size_t)j*CH + c] = xt[j][c];
  }
}

// ---------------------------------------------------------------------------
// Weight conversion to f16, transposed-B layout Wt[out_col][k]
// ---------------------------------------------------------------------------
__global__ __launch_bounds__(256) void k_wconv(
    const float* __restrict__ WQ, const float* __restrict__ WK,
    const float* __restrict__ WV, const float* __restrict__ WO,
    const float* __restrict__ F1, const float* __restrict__ F2,
    _Float16* __restrict__ wqkv, _Float16* __restrict__ wo,
    _Float16* __restrict__ w1,   _Float16* __restrict__ w2) {
  int idx = blockIdx.x * 256 + threadIdx.x;
  const int n_qkv = 3*CH*CH, n_wo = CH*CH, n_w1 = CFF*CH, n_w2 = CH*CFF;
  if (idx < n_qkv) {
    int col = idx / CH, c = idx % CH;
    int sel = col / CH, cc = col % CH, h = cc / HD, d = cc % HD;
    const float* s = sel == 0 ? WQ : (sel == 1 ? WK : WV);
    wqkv[idx] = (_Float16)s[(h*CH + c)*HD + d];       // Wt[h*32+d][c]
    return;
  }
  idx -= n_qkv;
  if (idx < n_wo) {
    int col = idx / CH, c = idx % CH;
    wo[idx] = (_Float16)WO[c*CH + col];               // Wt[c_out][c_in]
    return;
  }
  idx -= n_wo;
  if (idx < n_w1) { w1[idx] = (_Float16)F1[idx]; return; }
  idx -= n_w1;
  if (idx < n_w2) { w2[idx] = (_Float16)F2[idx]; return; }
}

// ---------------------------------------------------------------------------
// WMMA GEMM: C[N,M] = A[N,K] x Bt[M,K]^T  (f16 in, f32 acc)
// 6 waves / 192 threads, tile 96 rows x 96 cols, K step 32 (K compile-time).
// Each thread stages exactly 4 x 16B chunks per K-step; LDS double-buffered
// via GLOBAL_LOAD_ASYNC_TO_LDS_B128 + s_wait_asynccnt pipeline when available.
// compute(): preload all fragments (one dscnt drain), then 6 WMMAs back-to-back.
// EPI: 0 = f16 store, 1 = f16 bias+relu, 2 = f32 bias, 3 = f32 plain
// ---------------------------------------------------------------------------
template <int EPI, int K>
__global__ __launch_bounds__(192) void k_gemm(const _Float16* __restrict__ A,
                                              const _Float16* __restrict__ Bt,
                                              void* __restrict__ Cv,
                                              const float* __restrict__ bias,
                                              int M) {
#if ASYNC_OK
  __shared__ _Float16 As[2][96][40];   // pad 32->40 halves (conflict-free b128)
  __shared__ _Float16 Bs[2][96][40];
#else
  __shared__ _Float16 As[1][96][40];
  __shared__ _Float16 Bs[1][96][40];
#endif
  const int tid  = threadIdx.x;
  const int wave = tid >> 5;
  const int lane = tid & 31;
  const int hlf  = lane >> 4;
  const int lr   = lane & 15;
  const long rbase = (long)blockIdx.y * 96;
  const long cbase = (long)blockIdx.x * 96;

  // staging map: thread t covers rows (t>>2) and (t>>2)+48, 16B seg (t&3),
  // in both the A and the B slab -> 4 chunks/thread/stage.
  const int r0 = tid >> 2, s0 = tid & 3;
  const _Float16* pa0 = A  + (rbase + r0) * K + s0 * 8;
  const _Float16* pa1 = pa0 + 48 * K;
  const _Float16* pb0 = Bt + (cbase + r0) * K + s0 * 8;
  const _Float16* pb1 = pb0 + 48 * K;

  v8f acc[6] = {v8f{}, v8f{}, v8f{}, v8f{}, v8f{}, v8f{}};

  auto stage = [&](int buf, int kb) {
#if ASYNC_OK
    __builtin_amdgcn_global_load_async_to_lds_b128(
        AS1_V4I(pa0 + kb), AS3_V4I(&As[buf][r0     ][s0*8]), 0, 0);
    __builtin_amdgcn_global_load_async_to_lds_b128(
        AS1_V4I(pa1 + kb), AS3_V4I(&As[buf][r0 + 48][s0*8]), 0, 0);
    __builtin_amdgcn_global_load_async_to_lds_b128(
        AS1_V4I(pb0 + kb), AS3_V4I(&Bs[buf][r0     ][s0*8]), 0, 0);
    __builtin_amdgcn_global_load_async_to_lds_b128(
        AS1_V4I(pb1 + kb), AS3_V4I(&Bs[buf][r0 + 48][s0*8]), 0, 0);
#else
    v8h ra0 = *(const v8h*)(pa0 + kb);
    v8h ra1 = *(const v8h*)(pa1 + kb);
    v8h rb0 = *(const v8h*)(pb0 + kb);
    v8h rb1 = *(const v8h*)(pb1 + kb);
    *(v8h*)&As[buf][r0     ][s0*8] = ra0;
    *(v8h*)&As[buf][r0 + 48][s0*8] = ra1;
    *(v8h*)&Bs[buf][r0     ][s0*8] = rb0;
    *(v8h*)&Bs[buf][r0 + 48][s0*8] = rb1;
#endif
  };

  // fragment layout per ISA 7.12.2 (16-bit): K = e%8 + 8*(lane/16) + 16*(e/8)
  auto compute = [&](int buf) {
    const _Float16* bp = &Bs[buf][wave*16 + lr][hlf*8];
    v8h b0 = *(const v8h*)bp;
    v8h b1 = *(const v8h*)(bp + 16);
    v16h bf = __builtin_shufflevector(b0, b1, 0,1,2,3,4,5,6,7,8,9,10,11,12,13,14,15);
    v16h af[6];
#pragma unroll
    for (int m = 0; m < 6; ++m) {   // issue all LDS loads first (pipelined)
      const _Float16* ap = &As[buf][m*16 + lr][hlf*8];
      v8h a0 = *(const v8h*)ap;
      v8h a1 = *(const v8h*)(ap + 16);
      af[m] = __builtin_shufflevector(a0, a1, 0,1,2,3,4,5,6,7,8,9,10,11,12,13,14,15);
    }
#pragma unroll
    for (int m = 0; m < 6; ++m) {   // then 6 WMMAs back-to-back
      acc[m] = __builtin_amdgcn_wmma_f32_16x16x32_f16(
          false, af[m], false, bf, (short)0, acc[m], false, false);
    }
  };

  constexpr int NKS = K / 32;
#if ASYNC_OK
  stage(0, 0);
#pragma unroll 1
  for (int i = 0; i < NKS; ++i) {
    if (i + 1 < NKS) {
      stage((i + 1) & 1, (i + 1) * 32);   // issue next, then drain current
      WAIT_ASYNC(4);
    } else {
      WAIT_ASYNC(0);
    }
    __syncthreads();
    compute(i & 1);
    __syncthreads();
  }
#else
#pragma unroll 1
  for (int i = 0; i < NKS; ++i) {
    stage(0, i * 32);
    __syncthreads();
    compute(0);
    __syncthreads();
  }
#endif

  // epilogue: D layout — elem e of acc[m]: row = rbase + m*16 + 8*(lane/16)+e,
  //                                        col = cbase + wave*16 + lane%16
  const long col = cbase + wave*16 + lr;
  const float bv = (EPI == 1 || EPI == 2) ? bias[col] : 0.0f;
#pragma unroll
  for (int m = 0; m < 6; ++m) {
#pragma unroll
    for (int e = 0; e < 8; ++e) {
      long row = rbase + m*16 + hlf*8 + e;
      float v = acc[m][e] + bv;
      if (EPI == 1) v = v > 0.f ? v : 0.f;
      if (EPI == 0 || EPI == 1) ((_Float16*)Cv)[row*M + col] = (_Float16)v;
      else                      ((float*)Cv)[row*M + col] = v;
    }
  }
}

// ---------------------------------------------------------------------------
// 3x3 window attention: one thread per (token, head)
// ---------------------------------------------------------------------------
__global__ __launch_bounds__(256) void k_attn(const _Float16* __restrict__ qkv,
                                              _Float16* __restrict__ attn) {
  int t = blockIdx.x * 256 + threadIdx.x;
  if (t >= NT * NHEAD) return;
  int tok = t / NHEAD, h = t % NHEAD;
  int j = tok % WW, i = (tok / WW) % HH;
  _Float16* od = attn + (size_t)tok*CH + h*HD;
  if (i == 0 || i == HH-1 || j == 0 || j == WW-1) {
    v8h z{};
    *(v8h*)od = z; *(v8h*)(od+8) = z; *(v8h*)(od+16) = z; *(v8h*)(od+24) = z;
    return;
  }
  const _Float16* qp = qkv + (size_t)tok*864 + h*HD;
  float q[HD];
#pragma unroll
  for (int w = 0; w < 4; ++w) {
    v8h v = *(const v8h*)(qp + w*8);
#pragma unroll
    for (int e = 0; e < 8; ++e) q[w*8+e] = (float)v[e];
  }
  const float scale = 0.17677669529663687f;   // 1/sqrt(32)
  float sc[9];
#pragma unroll
  for (int n = 0; n < 9; ++n) {
    int tn = tok + (n/3 - 1)*WW + (n%3 - 1);
    const _Float16* kp = qkv + (size_t)tn*864 + CH + h*HD;
    float s = 0.f;
#pragma unroll
    for (int w = 0; w < 4; ++w) {
      v8h kv = *(const v8h*)(kp + w*8);
#pragma unroll
      for (int e = 0; e < 8; ++e) s += q[w*8+e] * (float)kv[e];
    }
    sc[n] = s * scale;
  }
  float mx = sc[0];
#pragma unroll
  for (int n = 1; n < 9; ++n) mx = fmaxf(mx, sc[n]);
  float den = 0.f;
#pragma unroll
  for (int n = 0; n < 9; ++n) { sc[n] = __expf(sc[n] - mx); den += sc[n]; }
  float inv = 1.f / den;
  float o[HD];
#pragma unroll
  for (int d = 0; d < HD; ++d) o[d] = 0.f;
#pragma unroll
  for (int n = 0; n < 9; ++n) {
    int tn = tok + (n/3 - 1)*WW + (n%3 - 1);
    const _Float16* vp = qkv + (size_t)tn*864 + 2*CH + h*HD;
    float p = sc[n] * inv;
#pragma unroll
    for (int w = 0; w < 4; ++w) {
      v8h vv = *(const v8h*)(vp + w*8);
#pragma unroll
      for (int e = 0; e < 8; ++e) o[w*8+e] += p * (float)vv[e];
    }
  }
#pragma unroll
  for (int w = 0; w < 4; ++w) {
    v8h ov;
#pragma unroll
    for (int e = 0; e < 8; ++e) ov[e] = (_Float16)o[w*8+e];
    *(v8h*)(od + w*8) = ov;
  }
}

// ---------------------------------------------------------------------------
__device__ inline float wave_sum(float v) {
#pragma unroll
  for (int m = 16; m >= 1; m >>= 1) v += __shfl_xor(v, m, 32);
  return v;
}

// LN1 + relu -> f16 (one wave per 288-channel row)
__global__ __launch_bounds__(256) void k_ln1(const float* __restrict__ Y,
                                             const float* __restrict__ g,
                                             const float* __restrict__ be,
                                             _Float16* __restrict__ out) {
  int row  = blockIdx.x * 8 + (threadIdx.x >> 5);
  int lane = threadIdx.x & 31;
  const float* yr = Y + (size_t)row * CH;
  float vals[9], s = 0.f, s2 = 0.f;
#pragma unroll
  for (int t = 0; t < 9; ++t) {
    float v = yr[t*32 + lane];
    vals[t] = v; s += v; s2 += v*v;
  }
  s = wave_sum(s); s2 = wave_sum(s2);
  float mean = s * (1.f/CH);
  float var  = s2 * (1.f/CH) - mean*mean;
  float rs   = rsqrtf(var + 1e-5f);
  _Float16* od = out + (size_t)row*CH;
#pragma unroll
  for (int t = 0; t < 9; ++t) {
    int c = t*32 + lane;
    float v = (vals[t] - mean)*rs*g[c] + be[c];
    od[c] = (_Float16)fmaxf(v, 0.f);
  }
}

// LN2 stats (mean, rstd) per token
__global__ __launch_bounds__(256) void k_ln2s(const float* __restrict__ Y,
                                              float* __restrict__ stats) {
  int row  = blockIdx.x * 8 + (threadIdx.x >> 5);
  int lane = threadIdx.x & 31;
  const float* yr = Y + (size_t)row * CH;
  float s = 0.f, s2 = 0.f;
#pragma unroll
  for (int t = 0; t < 9; ++t) {
    float v = yr[t*32 + lane];
    s += v; s2 += v*v;
  }
  s = wave_sum(s); s2 = wave_sum(s2);
  float mean = s * (1.f/CH);
  float var  = s2 * (1.f/CH) - mean*mean;
  if (lane == 0) {
    stats[row*2]     = mean;
    stats[row*2 + 1] = rsqrtf(var + 1e-5f);
  }
}

// Final: LN2-apply + residual(x) + relu + NHWC->NCHW (LDS-tiled, coalesced)
__global__ __launch_bounds__(192) void k_final(const float* __restrict__ Y2,
                                               const float* __restrict__ stats,
                                               const float* __restrict__ g,
                                               const float* __restrict__ be,
                                               const float* __restrict__ x,
                                               float* __restrict__ out) {
  __shared__ float yt[WW][CH + 1];
  __shared__ float sm[WW], sr[WW];
  int blk = blockIdx.x;
  int b = blk / HH, i = blk % HH;
  int tok0 = (b*HH + i) * WW;
  for (int idx = threadIdx.x; idx < WW*CH; idx += 192) {
    int j = idx / CH, c = idx % CH;
    yt[j][c] = Y2[(size_t)(tok0 + j)*CH + c];
  }
  if (threadIdx.x < WW) {
    sm[threadIdx.x] = stats[(tok0 + threadIdx.x)*2];
    sr[threadIdx.x] = stats[(tok0 + threadIdx.x)*2 + 1];
  }
  __syncthreads();
  int cg = threadIdx.x / WW;   // 0..3
  int j  = threadIdx.x % WW;
  for (int c = cg; c < CH; c += 4) {
    float v  = (yt[j][c] - sm[j]) * sr[j] * g[c] + be[c];
    size_t o = (((size_t)b*CH + c)*HH + i)*WW + j;
    out[o] = fmaxf(v + x[o], 0.f);
  }
}

// ---------------------------------------------------------------------------
extern "C" void kernel_launch(void* const* d_in, const int* in_sizes, int n_in,
                              void* d_out, int out_size, void* d_ws, size_t ws_size,
                              hipStream_t stream) {
  const float* x    = (const float*)d_in[0];
  const float* WQ   = (const float*)d_in[1];
  const float* WK   = (const float*)d_in[2];
  const float* WV   = (const float*)d_in[3];
  const float* WO   = (const float*)d_in[4];
  const float* f1w  = (const float*)d_in[5];
  const float* f1b  = (const float*)d_in[6];
  const float* f2w  = (const float*)d_in[7];
  const float* f2b  = (const float*)d_in[8];
  const float* ln1g = (const float*)d_in[9];
  const float* ln1b = (const float*)d_in[10];
  const float* ln2g = (const float*)d_in[11];
  const float* ln2b = (const float*)d_in[12];

  char* ws = (char*)d_ws;
  _Float16* xlh  = (_Float16*)(ws + O_XLH);
  _Float16* wqkv = (_Float16*)(ws + O_WQKV);
  _Float16* wo   = (_Float16*)(ws + O_WO);
  _Float16* w1   = (_Float16*)(ws + O_W1);
  _Float16* w2   = (_Float16*)(ws + O_W2);
  _Float16* qkv  = (_Float16*)(ws + O_QKV);
  _Float16* hin  = (_Float16*)(ws + O_QKV);   // alias: qkv dead after attention
  _Float16* attn = (_Float16*)(ws + O_ATTN);
  float*    y    = (float*)(ws + O_Y);
  float*    y2   = (float*)(ws + O_Y);        // alias: y dead after LN1
  _Float16* h1   = (_Float16*)(ws + O_H1);
  float*    st   = (float*)(ws + O_STAT);

  // 1) activations + weights -> f16
  k_xl<<<dim3(BT*HH), 256, 0, stream>>>(x, xlh);
  k_wconv<<<dim3(7128), 256, 0, stream>>>(WQ, WK, WV, WO, f1w, f2w, wqkv, wo, w1, w2);
  // 2) QKV projection: [NT,288] x [288,864]
  k_gemm<0,CH><<<dim3(864/96, NT/96), 192, 0, stream>>>(xlh, wqkv, (void*)qkv, nullptr, 864);
  // 3) 3x3 window attention
  k_attn<<<dim3((NT*NHEAD + 255)/256), 256, 0, stream>>>(qkv, attn);
  // 4) W_O projection -> f32
  k_gemm<3,CH><<<dim3(CH/96, NT/96), 192, 0, stream>>>(attn, wo, (void*)y, nullptr, CH);
  // 5) LN1 + relu -> f16
  k_ln1<<<dim3(NT/8), 256, 0, stream>>>(y, ln1g, ln1b, hin);
  // 6) FFN1 + bias + relu -> f16 [NT,2592]
  k_gemm<1,CH><<<dim3(CFF/96, NT/96), 192, 0, stream>>>(hin, w1, (void*)h1, f1b, CFF);
  // 7) FFN2 + bias -> f32 [NT,288]
  k_gemm<2,CFF><<<dim3(CH/96, NT/96), 192, 0, stream>>>(h1, w2, (void*)y2, f2b, CH);
  // 8) LN2 stats, then fused LN2 + residual + relu + transpose-out
  k_ln2s<<<dim3(NT/8), 256, 0, stream>>>(y2, st);
  k_final<<<dim3(BT*HH), 192, 0, stream>>>(y2, st, ln2g, ln2b, x, (float*)d_out);
}
